// ComplexRecurrentModel_27650999452275
// MI455X (gfx1250) — matrix operand order repaired and checked by reading, not compiled, not run on hardware
//
#include <hip/hip_runtime.h>
#include <hip/hip_bf16.h>
#include <math.h>

#define D_ 512
#define M_ 4
#define B_ 4
#define S_ 128
#define KS_ 32
#define STEPS_ 16
#define EPS_ 1e-6f
#define SCALE_ 0.04419417382415922f   // 512^-0.5

typedef __attribute__((ext_vector_type(16))) __bf16 v16bf;
typedef __attribute__((ext_vector_type(8)))  __bf16 v8bf;
typedef __attribute__((ext_vector_type(8)))  float  v8f;

__device__ __forceinline__ __bf16 f2bf(float f) {
    union { float f; unsigned u; } v; v.f = f;
    unsigned r = (v.u + 0x7FFFu + ((v.u >> 16) & 1u)) >> 16;   // RNE
    union { unsigned short s; __bf16 b; } o; o.s = (unsigned short)r;
    return o.b;
}

__device__ __forceinline__ float sigmoidf_(float x) { return 1.f / (1.f + expf(-x)); }

__device__ __forceinline__ v8f wmma_bf16(v16bf a, v16bf b, v8f c) {
    return __builtin_amdgcn_wmma_f32_16x16x32_bf16(false, a, false, b, (short)0, c, false, false);
}

// A fragment (16x32 bf16, row-major source): lane = M row; lanes 0-15 hold K 0-15,
// lanes 16-31 hold K 16-31.
__device__ __forceinline__ v16bf load_fragA(const __bf16* base, int ld, int row0, int k0) {
    int l = threadIdx.x & 31;
    const __bf16* p = base + (size_t)(row0 + (l & 15)) * ld + k0 + ((l >> 4) << 4);
    v8bf lo = *(const v8bf*)p;
    v8bf hi = *(const v8bf*)(p + 8);
    v16bf f;
#pragma unroll
    for (int i = 0; i < 8; ++i) { f[i] = lo[i]; f[i + 8] = hi[i]; }
    return f;
}

// B fragment (32x16 bf16) with B[k,n] = W[n0+n, k0+k] (W row-major, ld = row stride):
// lane = N col; lanes 0-15 hold K 0-15, lanes 16-31 hold K 16-31.
__device__ __forceinline__ v16bf load_fragB_rows(const __bf16* base, int ld, int k0, int n0) {
    int l = threadIdx.x & 31;
    const __bf16* p = base + (size_t)(n0 + (l & 15)) * ld + k0 + ((l >> 4) << 4);
    return *(const v16bf*)p;
}

// B fragment with B[k,n] = V[k0+k, n0+n] (V row-major): strided column gather.
__device__ __forceinline__ v16bf load_fragB_cols(const __bf16* base, int ld, int k0, int n0) {
    int l = threadIdx.x & 31;
    int n = n0 + (l & 15);
    int kb = k0 + ((l >> 4) << 4);
    v16bf f;
#pragma unroll
    for (int i = 0; i < 16; ++i) f[i] = base[(size_t)(kb + i) * ld + n];
    return f;
}

__device__ __forceinline__ float wave_sum32(float v) {
#pragma unroll
    for (int o = 16; o > 0; o >>= 1) v += __shfl_down(v, o, 32);
    return __shfl(v, 0, 32);
}

// ---------------- weight / state prep ----------------
__global__ void convw_kernel(const float* __restrict__ src, __bf16* __restrict__ dst) {
    int t = blockIdx.x * 256 + threadIdx.x;
    dst[t] = f2bf(src[t]);
}

__global__ void init_kernel(const float* __restrict__ xr, const float* __restrict__ xi,
                            float* sr, float* si, float* mem, float* ptrv,
                            float* acc, float* rem) {
    int t = blockIdx.x * 256 + threadIdx.x;              // covers 524288
    if (t < B_ * S_ * D_) { sr[t] = xr[t]; si[t] = xi[t]; }
    if (t < B_ * S_ * 2 * D_) acc[t] = 0.f;
    if (t < B_ * KS_ * 2 * D_) mem[t] = 0.f;
    if (t < B_ * KS_) ptrv[t] = ((t & (KS_ - 1)) == 0) ? 1.f : 0.f;
    if (t < B_) rem[t] = 1.f;
}

__global__ void prep_kernel(const float* __restrict__ xr, const float* __restrict__ xi,
                            const float* __restrict__ sr, const float* __restrict__ si,
                            __bf16* __restrict__ cr, __bf16* __restrict__ ci) {
    int t = blockIdx.x * 256 + threadIdx.x;
    cr[t] = f2bf(0.5f * (xr[t] + sr[t]));
    ci[t] = f2bf(0.5f * (xi[t] + si[t]));
}

// ---------------- complex GEMM: z[b,m,s,o] = sum_d A[row,d] * W[m,o,d] (complex) ---------
__global__ void clin_gemm(const __bf16* __restrict__ Ar, const __bf16* __restrict__ Ai,
                          const __bf16* __restrict__ Wr, const __bf16* __restrict__ Wi,
                          float* __restrict__ Zr, float* __restrict__ Zi,
                          __bf16* __restrict__ Zrb, __bf16* __restrict__ Zib,
                          int a_per_m) {
    int wg = blockIdx.x * 8 + (threadIdx.x >> 5);         // 4096 wave-tiles
    int nt = wg & 31, st = (wg >> 5) & 7, m = (wg >> 8) & 3, b = wg >> 10;
    int n0 = nt * 16, s0 = st * 16;
    int arow0 = a_per_m ? ((b * M_ + m) * S_ + s0) : (b * S_ + s0);
    const __bf16* wr = Wr + (size_t)m * D_ * D_;
    const __bf16* wi = Wi + (size_t)m * D_ * D_;
    v8f arr = {}, aii = {}, air = {}, ari = {};
    for (int k0 = 0; k0 < D_; k0 += 32) {
        v16bf a_r = load_fragA(Ar, D_, arow0, k0);
        v16bf a_i = load_fragA(Ai, D_, arow0, k0);
        v16bf b_r = load_fragB_rows(wr, D_, k0, n0);
        v16bf b_i = load_fragB_rows(wi, D_, k0, n0);
        arr = wmma_bf16(a_r, b_r, arr);
        aii = wmma_bf16(a_i, b_i, aii);
        air = wmma_bf16(a_i, b_r, air);
        ari = wmma_bf16(a_r, b_i, ari);
    }
    int l = threadIdx.x & 31;
    int col = n0 + (l & 15);
    int r0 = (b * M_ + m) * S_ + s0 + ((l >> 4) << 3);
#pragma unroll
    for (int i = 0; i < 8; ++i) {
        float zr = arr[i] - aii[i];
        float zi = air[i] + ari[i];
        size_t idx = (size_t)(r0 + i) * D_ + col;
        if (Zr)  { Zr[idx] = zr;        Zi[idx] = zi; }
        if (Zrb) { Zrb[idx] = f2bf(zr); Zib[idx] = f2bf(zi); }
    }
}

// ---------------- magnitude-LN / phase / ReLU-mod nonlinearity -------------------------
__global__ void modnorm(const float* __restrict__ Zr, const float* __restrict__ Zi,
                        const float* __restrict__ ln_scale, const float* __restrict__ ln_shift,
                        const float* __restrict__ mod_bias,
                        __bf16* __restrict__ Zrb, __bf16* __restrict__ Zib) {
    int row = blockIdx.x;                                  // (b*M+m)*S + s
    int m = (row / S_) % M_;
    size_t base = (size_t)row * D_;
    int t = threadIdx.x;
    float zr0 = Zr[base + t],       zi0 = Zi[base + t];
    float zr1 = Zr[base + t + 256], zi1 = Zi[base + t + 256];
    float m0 = sqrtf(zr0 * zr0 + zi0 * zi0) + EPS_;
    float m1 = sqrtf(zr1 * zr1 + zi1 * zi1) + EPS_;
    __shared__ float red[256];
    red[t] = m0 + m1; __syncthreads();
    for (int o = 128; o > 0; o >>= 1) { if (t < o) red[t] += red[t + o]; __syncthreads(); }
    float mean = red[0] / (float)D_;
    __syncthreads();
    red[t] = m0 * m0 + m1 * m1; __syncthreads();
    for (int o = 128; o > 0; o >>= 1) { if (t < o) red[t] += red[t + o]; __syncthreads(); }
    float var = (red[0] - (float)D_ * mean * mean) / (float)(D_ - 1);
    float rstd = rsqrtf(var + EPS_);
#pragma unroll
    for (int j = 0; j < 2; ++j) {
        int e = t + j * 256;
        float zr = j ? zr1 : zr0, zi = j ? zi1 : zi0, mg = j ? m1 : m0;
        float nm = (mg - mean) * rstd * ln_scale[m * D_ + e] + ln_shift[m * D_ + e];
        float ph = atan2f(zi, zr);
        float nr = nm * cosf(ph), ni = nm * sinf(ph);
        float nrm = sqrtf(nr * nr + ni * ni) + EPS_;
        float sc = fmaxf(nrm + mod_bias[m * D_ + e], 0.f) / nrm;
        Zrb[base + e] = f2bf(nr * sc);
        Zib[base + e] = f2bf(ni * sc);
    }
}

// ---------------- fused attention per (b,m): scores -> softmax -> AV (+gate) ----------
__global__ void attn_kernel(const __bf16* __restrict__ qr, const __bf16* __restrict__ qi,
                            const __bf16* __restrict__ kr, const __bf16* __restrict__ ki,
                            const __bf16* __restrict__ vr, const __bf16* __restrict__ vi,
                            const float* __restrict__ gate_mask,
                            float* __restrict__ pr, float* __restrict__ pi) {
    int bx = blockIdx.x;
    int st = bx & 7, m = (bx >> 3) & 3, b = bx >> 5;
    int s0 = st * 16;
    size_t mat = (size_t)(b * M_ + m) * S_ * D_;
    const __bf16* Q_r = qr + mat; const __bf16* Q_i = qi + mat;
    const __bf16* K_r = kr + mat; const __bf16* K_i = ki + mat;
    const __bf16* V_r = vr + mat; const __bf16* V_i = vi + mat;
    __shared__ float sc[16][128];
    __shared__ alignas(32) __bf16 attnb[16][128];
    int w = threadIdx.x >> 5;
    int l = threadIdx.x & 31;
    {   // scores: qf . kf^T over K = 2D (real + imag halves)
        int t0 = w * 16;
        v8f acc = {};
        for (int k0 = 0; k0 < D_; k0 += 32)
            acc = wmma_bf16(load_fragA(Q_r, D_, s0, k0), load_fragB_rows(K_r, D_, k0, t0), acc);
        for (int k0 = 0; k0 < D_; k0 += 32)
            acc = wmma_bf16(load_fragA(Q_i, D_, s0, k0), load_fragB_rows(K_i, D_, k0, t0), acc);
        int col = t0 + (l & 15);
        int r0 = (l >> 4) << 3;
#pragma unroll
        for (int i = 0; i < 8; ++i) sc[r0 + i][col] = acc[i] * SCALE_;
    }
    __syncthreads();
    if (threadIdx.x < 16) {   // row softmax
        int r = threadIdx.x;
        float mx = -1e30f;
        for (int c = 0; c < 128; ++c) mx = fmaxf(mx, sc[r][c]);
        float s = 0.f;
        for (int c = 0; c < 128; ++c) { float e = expf(sc[r][c] - mx); sc[r][c] = e; s += e; }
        float inv = 1.f / s;
        for (int c = 0; c < 128; ++c) attnb[r][c] = f2bf(sc[r][c] * inv);
    }
    __syncthreads();
    {   // AV (real & imag) with gate fused in epilogue
        v8f accR[4] = {{}, {}, {}, {}};
        v8f accI[4] = {{}, {}, {}, {}};
        const __bf16* A = &attnb[0][0];
        for (int k0 = 0; k0 < 128; k0 += 32) {
            v16bf a = load_fragA(A, 128, 0, k0);
#pragma unroll
            for (int j = 0; j < 4; ++j) {
                int n0 = (w * 4 + j) * 16;
                accR[j] = wmma_bf16(a, load_fragB_cols(V_r, D_, k0, n0), accR[j]);
                accI[j] = wmma_bf16(a, load_fragB_cols(V_i, D_, k0, n0), accI[j]);
            }
        }
#pragma unroll
        for (int j = 0; j < 4; ++j) {
            int n0 = (w * 4 + j) * 16;
            int col = n0 + (l & 15);
            float g = sigmoidf_(gate_mask[m * D_ + col]);
            int r0 = s0 + ((l >> 4) << 3);
#pragma unroll
            for (int i = 0; i < 8; ++i) {
                size_t idx = mat + (size_t)(r0 + i) * D_ + col;
                pr[idx] = accR[j][i] * g;
                pi[idx] = accI[j][i] * g;
            }
        }
    }
}

// ---------------- per-(b,m,s) score/conf/halt heads -----------------------------------
__global__ void heads_kernel(const float* __restrict__ pr, const float* __restrict__ pi,
                             const float* __restrict__ score_w, const float* __restrict__ score_b,
                             const float* __restrict__ conf_w, const float* __restrict__ conf_b,
                             const float* __restrict__ halt_w, const float* __restrict__ halt_b,
                             float* __restrict__ score, float* __restrict__ conf,
                             float* __restrict__ halt_sig) {
    int bx = blockIdx.x;                                  // (b*M+m)*S + s
    int m = (bx / S_) % M_;
    size_t base = (size_t)bx * D_;
    int t = threadIdx.x;
    float as = 0.f, ac = 0.f, ah = 0.f;
    for (int e = t; e < 2 * D_; e += 128) {
        float f = (e < D_) ? pr[base + e] : pi[base + e - D_];
        as += f * score_w[m * 2 * D_ + e];
        ac += f * conf_w[m * 2 * D_ + e];
        ah += f * halt_w[m * 2 * D_ + e];
    }
    __shared__ float r0[128], r1[128], r2[128];
    r0[t] = as; r1[t] = ac; r2[t] = ah; __syncthreads();
    for (int o = 64; o > 0; o >>= 1) {
        if (t < o) { r0[t] += r0[t + o]; r1[t] += r1[t + o]; r2[t] += r2[t + o]; }
        __syncthreads();
    }
    if (t == 0) {
        score[bx] = r0[0] + score_b[m];
        conf[bx] = sigmoidf_(r1[0] + conf_b[m]);
        halt_sig[bx] = sigmoidf_(r2[0] + halt_b[m]);
    }
}

__global__ void p_reduce(const float* __restrict__ halt_sig, float* __restrict__ p) {
    int b = blockIdx.x, t = threadIdx.x;
    __shared__ float red[256];
    red[t] = halt_sig[b * M_ * S_ + t] + halt_sig[b * M_ * S_ + t + 256];
    __syncthreads();
    for (int o = 128; o > 0; o >>= 1) { if (t < o) red[t] += red[t + o]; __syncthreads(); }
    if (t == 0) p[b] = red[0] / (float)(M_ * S_);
}

__global__ void flatin_kernel(const float* __restrict__ sr, const float* __restrict__ si,
                              float* __restrict__ flat_in) {
    int t = blockIdx.x * 256 + threadIdx.x;               // b*1024 + e
    int b = t >> 10, e = t & 1023;
    const float* src = (e < D_) ? sr : si;
    int d = e & (D_ - 1);
    float s = 0.f;
    for (int ss = 0; ss < S_; ++ss) s += src[(size_t)(b * S_ + ss) * D_ + d];
    flat_in[t] = s / (float)S_;
}

// ---------------- controller + differentiable-stack pointer (one wave per batch) -------
__global__ void ctrl_stack_kernel(const float* __restrict__ flat_in,
                                  const float* __restrict__ ctrl_w, const float* __restrict__ ctrl_b,
                                  float* __restrict__ ptrv, float* __restrict__ wmask) {
    int b = threadIdx.x >> 5, l = threadIdx.x & 31;
    float c[3];
#pragma unroll
    for (int j = 0; j < 3; ++j) {
        float a = 0.f;
        for (int e = l; e < 2 * D_; e += 32) a += flat_in[b * 2 * D_ + e] * ctrl_w[e * 3 + j];
        c[j] = wave_sum32(a) + ctrl_b[j];
    }
    float mx = fmaxf(c[0], fmaxf(c[1], c[2]));
    float e0 = expf(c[0] - mx), e1 = expf(c[1] - mx), e2 = expf(c[2] - mx);
    float inv = 1.f / (e0 + e1 + e2);
    float push = e0 * inv, pop = e1 * inv, noop = e2 * inv;
    float pc = ptrv[b * KS_ + l];
    float up = __shfl(pc, (l + 31) & 31, 32);             // roll(+1)
    float dn = __shfl(pc, (l + 1) & 31, 32);              // roll(-1)
    float np = push * up + pop * dn + noop * pc;
    float sum = wave_sum32(np);
    np = np / (sum + EPS_);
    ptrv[b * KS_ + l] = np;
    wmask[b * KS_ + l] = push * up;
}

__global__ void memupd_kernel(const float* __restrict__ flat_in, const float* __restrict__ wmask,
                              const float* __restrict__ ptrv, float* __restrict__ mem,
                              float* __restrict__ readv) {
    int t = blockIdx.x * 256 + threadIdx.x;               // b*1024+e (4096)
    int b = t >> 10;
    float fin = flat_in[t];
    float rd = 0.f;
    for (int k = 0; k < KS_; ++k) {
        float wv = wmask[b * KS_ + k];
        size_t idx = (size_t)(b * KS_ + k) * 1024 + (t & 1023);
        float nm = wv * fin + mem[idx] * (1.f - wv);
        mem[idx] = nm;
        rd += nm * ptrv[b * KS_ + k];
    }
    readv[t] = rd;
}

__global__ void st_kernel(const float* __restrict__ readv,
                          const float* __restrict__ st_score_w, const float* __restrict__ st_score_b,
                          const float* __restrict__ st_conf_w, const float* __restrict__ st_conf_b,
                          float* __restrict__ st_sc, float* __restrict__ st_cf) {
    int b = threadIdx.x >> 5, l = threadIdx.x & 31;
    float a = 0.f, c = 0.f;
    for (int e = l; e < 2 * D_; e += 32) {
        float r = readv[b * 2 * D_ + e];
        a += r * st_score_w[e];
        c += r * st_conf_w[e];
    }
    a = wave_sum32(a); c = wave_sum32(c);
    if (l == 0) { st_sc[b] = a + st_score_b[0]; st_cf[b] = sigmoidf_(c + st_conf_b[0]); }
}

// ---------------- combine over M+1 streams + ACT halting accumulation ------------------
__global__ void combine_kernel(const float* __restrict__ score, const float* __restrict__ conf,
                               const float* __restrict__ st_sc, const float* __restrict__ st_cf,
                               const float* __restrict__ pr, const float* __restrict__ pi,
                               const float* __restrict__ readv,
                               const float* __restrict__ p, const float* __restrict__ rem,
                               float* __restrict__ sr, float* __restrict__ si,
                               float* __restrict__ acc) {
    int bs = blockIdx.x;
    int b = bs / S_, s = bs % S_;
    float lg[5];
#pragma unroll
    for (int m = 0; m < M_; ++m) {
        int idx = (b * M_ + m) * S_ + s;
        lg[m] = score[idx] * conf[idx];
    }
    lg[4] = st_sc[b] * st_cf[b];
    float mx = lg[0];
#pragma unroll
    for (int j = 1; j < 5; ++j) mx = fmaxf(mx, lg[j]);
    float ws[5], sum = 0.f;
#pragma unroll
    for (int j = 0; j < 5; ++j) { ws[j] = expf(lg[j] - mx); sum += ws[j]; }
    float inv = 1.f / sum;
#pragma unroll
    for (int j = 0; j < 5; ++j) ws[j] *= inv;
    float pf = p[b] * rem[b];
    for (int e = threadIdx.x; e < 2 * D_; e += 256) {
        int d = e & (D_ - 1);
        const float* src = (e < D_) ? pr : pi;
        float v = ws[4] * readv[b * 2 * D_ + e];
#pragma unroll
        for (int m = 0; m < M_; ++m)
            v += ws[m] * src[((size_t)(b * M_ + m) * S_ + s) * D_ + d];
        if (e < D_) sr[(size_t)bs * D_ + d] = v;
        else        si[(size_t)bs * D_ + d] = v;
        acc[(size_t)bs * 2 * D_ + e] += pf * v;
    }
}

__global__ void rem_upd(const float* __restrict__ p, float* __restrict__ rem) {
    int t = threadIdx.x;
    if (t < B_) rem[t] *= (1.f - p[t]);
}

__global__ void final_kernel(const float* __restrict__ acc, const float* __restrict__ rem,
                             const float* __restrict__ sr, const float* __restrict__ si,
                             float* __restrict__ out) {
    int t = blockIdx.x * 256 + threadIdx.x;               // 524288
    int b = t >> 17;
    int row = t >> 10;
    int e = t & 1023;
    float v = (e < D_) ? sr[(size_t)row * D_ + e] : si[(size_t)row * D_ + (e - D_)];
    out[t] = acc[t] + rem[b] * v;
}

extern "C" void kernel_launch(void* const* d_in, const int* in_sizes, int n_in,
                              void* d_out, int out_size, void* d_ws, size_t ws_size,
                              hipStream_t stream) {
    (void)in_sizes; (void)n_in; (void)out_size; (void)ws_size;
    const float* x_real = (const float*)d_in[0];
    const float* x_imag = (const float*)d_in[1];
    const float* wsrc[8] = { (const float*)d_in[2], (const float*)d_in[3],
                             (const float*)d_in[4], (const float*)d_in[5],
                             (const float*)d_in[6], (const float*)d_in[7],
                             (const float*)d_in[8], (const float*)d_in[9] };
    const float* ln_scale = (const float*)d_in[10];
    const float* ln_shift = (const float*)d_in[11];
    const float* mod_bias = (const float*)d_in[12];
    const float* gate_mask = (const float*)d_in[13];
    const float* score_w = (const float*)d_in[14];
    const float* score_b = (const float*)d_in[15];
    const float* conf_w = (const float*)d_in[16];
    const float* conf_b = (const float*)d_in[17];
    const float* halt_w = (const float*)d_in[18];
    const float* halt_b = (const float*)d_in[19];
    const float* ctrl_w = (const float*)d_in[20];
    const float* ctrl_b = (const float*)d_in[21];
    const float* st_score_w = (const float*)d_in[22];
    const float* st_score_b = (const float*)d_in[23];
    const float* st_conf_w = (const float*)d_in[24];
    const float* st_conf_b = (const float*)d_in[25];

    size_t off = 0;
    auto get = [&](size_t bytes) -> void* {
        void* p = (char*)d_ws + off;
        off += (bytes + 255) & ~(size_t)255;
        return p;
    };
    const size_t WEL  = (size_t)M_ * D_ * D_;             // 1,048,576
    const size_t NSD  = (size_t)B_ * S_ * D_;             // 262,144
    const size_t NMSD = (size_t)B_ * M_ * S_ * D_;        // 1,048,576

    __bf16* wb[8];
    for (int i = 0; i < 8; ++i) wb[i] = (__bf16*)get(WEL * 2);
    float* sr  = (float*)get(NSD * 4);
    float* si  = (float*)get(NSD * 4);
    __bf16* cr = (__bf16*)get(NSD * 2);
    __bf16* ci = (__bf16*)get(NSD * 2);
    float* zr  = (float*)get(NMSD * 4);
    float* zi  = (float*)get(NMSD * 4);
    __bf16* zbr = (__bf16*)get(NMSD * 2);
    __bf16* zbi = (__bf16*)get(NMSD * 2);
    __bf16* qbr = (__bf16*)get(NMSD * 2);
    __bf16* qbi = (__bf16*)get(NMSD * 2);
    __bf16* kbr = (__bf16*)get(NMSD * 2);
    __bf16* kbi = (__bf16*)get(NMSD * 2);
    __bf16* vbr = (__bf16*)get(NMSD * 2);
    __bf16* vbi = (__bf16*)get(NMSD * 2);
    float* pr    = (float*)get(NMSD * 4);
    float* pi    = (float*)get(NMSD * 4);
    float* score = (float*)get((size_t)B_ * M_ * S_ * 4);
    float* conf  = (float*)get((size_t)B_ * M_ * S_ * 4);
    float* halt  = (float*)get((size_t)B_ * M_ * S_ * 4);
    float* flatin = (float*)get((size_t)B_ * 2 * D_ * 4);
    float* ptrv   = (float*)get((size_t)B_ * KS_ * 4);
    float* wmask  = (float*)get((size_t)B_ * KS_ * 4);
    float* memb   = (float*)get((size_t)B_ * KS_ * 2 * D_ * 4);
    float* readv  = (float*)get((size_t)B_ * 2 * D_ * 4);
    float* stsc   = (float*)get(256);
    float* stcf   = (float*)get(256);
    float* pbuf   = (float*)get(256);
    float* rembuf = (float*)get(256);
    float* accb   = (float*)get((size_t)B_ * S_ * 2 * D_ * 4);

    for (int i = 0; i < 8; ++i)
        convw_kernel<<<4096, 256, 0, stream>>>(wsrc[i], wb[i]);
    init_kernel<<<2048, 256, 0, stream>>>(x_real, x_imag, sr, si, memb, ptrv, accb, rembuf);

    for (int step = 0; step < STEPS_; ++step) {
        prep_kernel<<<1024, 256, 0, stream>>>(x_real, x_imag, sr, si, cr, ci);
        clin_gemm<<<512, 256, 0, stream>>>(cr, ci, wb[0], wb[1], zr, zi, nullptr, nullptr, 0);
        modnorm<<<2048, 256, 0, stream>>>(zr, zi, ln_scale, ln_shift, mod_bias, zbr, zbi);
        clin_gemm<<<512, 256, 0, stream>>>(zbr, zbi, wb[2], wb[3], nullptr, nullptr, qbr, qbi, 1);
        clin_gemm<<<512, 256, 0, stream>>>(zbr, zbi, wb[4], wb[5], nullptr, nullptr, kbr, kbi, 1);
        clin_gemm<<<512, 256, 0, stream>>>(zbr, zbi, wb[6], wb[7], nullptr, nullptr, vbr, vbi, 1);
        attn_kernel<<<128, 256, 0, stream>>>(qbr, qbi, kbr, kbi, vbr, vbi, gate_mask, pr, pi);
        heads_kernel<<<2048, 128, 0, stream>>>(pr, pi, score_w, score_b, conf_w, conf_b,
                                               halt_w, halt_b, score, conf, halt);
        p_reduce<<<4, 256, 0, stream>>>(halt, pbuf);
        flatin_kernel<<<16, 256, 0, stream>>>(sr, si, flatin);
        ctrl_stack_kernel<<<1, 128, 0, stream>>>(flatin, ctrl_w, ctrl_b, ptrv, wmask);
        memupd_kernel<<<16, 256, 0, stream>>>(flatin, wmask, ptrv, memb, readv);
        st_kernel<<<1, 128, 0, stream>>>(readv, st_score_w, st_score_b, st_conf_w, st_conf_b,
                                         stsc, stcf);
        combine_kernel<<<512, 256, 0, stream>>>(score, conf, stsc, stcf, pr, pi, readv,
                                                pbuf, rembuf, sr, si, accb);
        rem_upd<<<1, 32, 0, stream>>>(pbuf, rembuf);
    }
    final_kernel<<<2048, 256, 0, stream>>>(accb, rembuf, sr, si, (float*)d_out);
}
